// SuperAggressiveAttention_89850715832914
// MI455X (gfx1250) — matrix operand order repaired and checked by reading, not compiled, not run on hardware
//
#include <hip/hip_runtime.h>
#include <stdint.h>

// ---------------------------------------------------------------------------
// SuperAggressiveAttention for MI455X (gfx1250, wave32, WMMA bf16).
//
// All three similarity components derive from ONE dots GEMM:
//   cos  = clip(dots * rn1[n] * rn2[m])              (reciprocal norms)
//   cov  = (dots - a[n] - mq[n]*ksum[m] + mq[n]*c)/sqrt(64)  (rank-1 corr.)
//   csim = clip(dots * rqd[n] * rkd[m]) -> margin -> var_c (row broadcast)
// Global stds gate scalars -> 3 passes over the score tensor, each
// recomputing the cheap bf16 WMMA GEMM (FQ/FK/FVT live in L2) instead of
// spilling 134MB score tensors to HBM.  All per-element divides are
// replaced by precomputed separable reciprocals (divide-free hot loops).
// ---------------------------------------------------------------------------

#define H      8
#define QB     4
#define NTOK   1024
#define DIM    512
#define DHEAD  64
#define INNER  512
#define NROWS  (QB * NTOK)   // 4096
#define HQ     (H * QB)      // 32
#define GAMMA_C 0.01f

typedef __attribute__((ext_vector_type(16))) __bf16 v16bf;
typedef __attribute__((ext_vector_type(8)))  __bf16 v8bf;
typedef __attribute__((ext_vector_type(8)))  float  v8f;

union Frag { v16bf v; v8bf h[2]; };

static __device__ inline unsigned short f2bf(float f) {
  unsigned u = __builtin_bit_cast(unsigned, f);
  u += 0x7FFFu + ((u >> 16) & 1u);          // round-to-nearest-even
  return (unsigned short)(u >> 16);
}
static __device__ inline float bf2f(unsigned short h) {
  unsigned u = ((unsigned)h) << 16;
  return __builtin_bit_cast(float, u);
}
static __device__ inline float wave_sum(float x) {
#pragma unroll
  for (int m = 1; m < 32; m <<= 1) x += __shfl_xor(x, m, 32);
  return x;
}
static __device__ inline float half_sum(float x) {   // within 16-lane half
#pragma unroll
  for (int m = 1; m < 16; m <<= 1) x += __shfl_xor(x, m, 32);
  return x;
}
static __device__ inline float half_max(float x) {
#pragma unroll
  for (int m = 1; m < 16; m <<= 1) x = fmaxf(x, __shfl_xor(x, m, 32));
  return x;
}
static __device__ inline v8f wmma_bf(const Frag& a, const Frag& b, v8f c) {
  return __builtin_amdgcn_wmma_f32_16x16x32_bf16(false, a.v, false, b.v,
                                                 (short)0, c, false, false);
}

// ---------------- init / weight transpose ----------------------------------
__global__ void k_zero(float* SC, float* QGS, float* KGS) {
  int t = threadIdx.x;
  if (t < 256) SC[t] = 0.f;
  for (int i = t; i < 512; i += 256) { QGS[i] = 0.f; KGS[i] = 0.f; }
}

__global__ void k_transpose_w(const float* W, unsigned short* WT) {
  int idx = blockIdx.x * blockDim.x + threadIdx.x;   // 512*512
  int i = idx >> 9, j = idx & 511;                   // W[i][j]
  WT[j * 512 + i] = f2bf(W[idx]);
}

// ---------------- LayerNorm -> bf16 -----------------------------------------
__global__ void k_layernorm(const float* __restrict__ X, const float* __restrict__ g,
                            const float* __restrict__ bta, unsigned short* __restrict__ Y) {
  int wave = threadIdx.x >> 5, lane = threadIdx.x & 31;
  int row = blockIdx.x * 4 + wave;
  const float* xr = X + (size_t)row * DIM;
  float xs[16], s = 0.f, s2 = 0.f;
#pragma unroll
  for (int i = 0; i < 16; i++) { float x = xr[lane + 32 * i]; xs[i] = x; s += x; s2 += x * x; }
  s = wave_sum(s); s2 = wave_sum(s2);
  float mu = s * (1.f / DIM);
  float rs = rsqrtf(s2 * (1.f / DIM) - mu * mu + 1e-5f);
  unsigned short* yr = Y + (size_t)row * DIM;
#pragma unroll
  for (int i = 0; i < 16; i++) {
    int j = lane + 32 * i;
    yr[j] = f2bf((xs[i] - mu) * rs * g[j] + bta[j]);
  }
}

// ---------------- projection GEMM: LN(x)[4096x512] @ WinT -------------------
__global__ void k_proj_gemm(const unsigned short* __restrict__ XLN,
                            const unsigned short* __restrict__ WT,
                            unsigned short* __restrict__ FQ,
                            unsigned short* __restrict__ FK,
                            unsigned short* __restrict__ FVT, int mode) {
  int wave = threadIdx.x >> 5, lane = threadIdx.x & 31;
  int n16 = lane & 15, hi = lane >> 4;
  int tile = blockIdx.x * 4 + wave;       // 8192 tiles
  int mt = tile >> 5, jt = tile & 31;
  const unsigned short* ar = XLN + (size_t)(mt * 16 + n16) * DIM;
  const unsigned short* br = WT + (size_t)(jt * 16 + n16) * DIM;
  v8f c = {};
#pragma unroll 4
  for (int kb = 0; kb < DIM; kb += 32) {
    Frag a, b;
    a.h[0] = *(const v8bf*)(ar + kb + 16 * hi);
    a.h[1] = *(const v8bf*)(ar + kb + 8 + 16 * hi);
    b.v    = *(const v16bf*)(br + kb + 16 * hi);
    c = wmma_bf(a, b, c);
  }
#pragma unroll
  for (int r = 0; r < 8; r++) {
    int gr = mt * 16 + r + 8 * hi;          // token row
    int col = jt * 16 + n16;                // inner col
    int h = col >> 6, d = col & 63, q = gr >> 10, n = gr & 1023;
    unsigned short v = f2bf(c[r]);
    if (mode == 0)      FQ[(size_t)((h * QB + q) * NTOK + n) * DHEAD + d] = v;
    else if (mode == 1) FK[(size_t)((h * QB + q) * NTOK + n) * DHEAD + d] = v;
    else                FVT[(size_t)((h * QB + q) * DHEAD + d) * NTOK + n] = v;
  }
}

// ---------------- per-row stats (reciprocals!) + column sums ----------------
// q-mode: OUT0 = 1/(||row||+1e-8), OUT1 = 1/max(||row||,1e-6), OUT2 = mean_d
// k-mode: OUT0 = 1/(||row||+1e-8), OUT1 = 1/max(||row||,1e-6), OUT2 = sum_d
__global__ void k_rowstats(const unsigned short* __restrict__ F, float* OUT0, float* OUT1,
                           float* OUT2, float* COLG, float* MKB, int is_k) {
  __shared__ float colsum[64];
  int tid = threadIdx.x, wave = tid >> 5, lane = tid & 31;
  int hq = blockIdx.x;
  if (tid < 64) colsum[tid] = 0.f;
  __syncthreads();
  const unsigned short* base = F + (size_t)hq * NTOK * DHEAD;
  float cs0 = 0.f, cs1 = 0.f;
  for (int rr = wave; rr < NTOK; rr += 8) {
    const unsigned short* row = base + (size_t)rr * DHEAD;
    float x0 = bf2f(row[lane]), x1 = bf2f(row[lane + 32]);
    float s  = wave_sum(x0 + x1);
    float s2 = wave_sum(x0 * x0 + x1 * x1);
    cs0 += x0; cs1 += x1;
    if (lane == 0) {
      float nrm = sqrtf(s2);
      int g = hq * NTOK + rr;
      OUT0[g] = 1.f / (nrm + 1e-8f);
      OUT1[g] = 1.f / fmaxf(nrm, 1e-6f);
      OUT2[g] = is_k ? s : s * (1.f / DHEAD);   // ksum[m] or mq[n]
    }
  }
  atomicAdd(&colsum[lane], cs0);
  atomicAdd(&colsum[lane + 32], cs1);
  __syncthreads();
  if (tid < 64) {
    float tot = colsum[tid];
    int h = hq >> 2;
    if (is_k) { MKB[hq * DHEAD + tid] = tot * (1.f / NTOK); atomicAdd(&COLG[h * DHEAD + tid], tot); }
    else      { atomicAdd(&COLG[h * DHEAD + tid], tot); }
  }
}

// a[n] = f_q[n] . mkbar
__global__ void k_compute_a(const unsigned short* __restrict__ FQ,
                            const float* __restrict__ MKB, float* AV) {
  int wave = threadIdx.x >> 5, lane = threadIdx.x & 31;
  int row = blockIdx.x * 4 + wave;          // 0..32767
  int hq = row >> 10;
  const unsigned short* r = FQ + (size_t)row * DHEAD;
  const float* mk = MKB + hq * DHEAD;
  float d = bf2f(r[lane]) * mk[lane] + bf2f(r[lane + 32]) * mk[lane + 32];
  d = wave_sum(d);
  if (lane == 0) AV[row] = d;
}

// ---------------- tiny weight-predictor MLP (1 workgroup) -------------------
__global__ void k_mlp(const float* QGS, const float* KGS, const float* MKB,
                      const float* W1, const float* b1, const float* lng, const float* lnb,
                      const float* W2, const float* b2, const float* W3, const float* b3,
                      const float* wtemp, float* SC) {
  __shared__ float feats[H][128];
  __shared__ float h1[H][64];
  __shared__ float h2[H][32];
  __shared__ float mv[H][2];
  int tid = threadIdx.x;
  if (tid < HQ) {                          // c[h][q] = sum_d mkbar
    float s = 0.f;
    for (int d = 0; d < DHEAD; d++) s += MKB[tid * DHEAD + d];
    SC[64 + tid] = s;
  }
  for (int i = tid; i < H * 128; i += 256) {
    int h = i >> 7, j = i & 127;
    feats[h][j] = (j < 64 ? QGS[h * 64 + j] : KGS[h * 64 + j - 64]) * (1.f / (float)NROWS);
  }
  __syncthreads();
  for (int i = tid; i < H * 64; i += 256) {
    int h = i >> 6, j = i & 63;
    float s = b1[j];
    for (int kk = 0; kk < 128; kk++) s += feats[h][kk] * W1[kk * 64 + j];
    h1[h][j] = s;
  }
  __syncthreads();
  if (tid < H) {
    float s = 0.f, s2 = 0.f;
    for (int j = 0; j < 64; j++) { float x = h1[tid][j]; s += x; s2 += x * x; }
    float mu = s / 64.f;
    mv[tid][0] = mu; mv[tid][1] = rsqrtf(s2 / 64.f - mu * mu + 1e-5f);
  }
  __syncthreads();
  for (int i = tid; i < H * 64; i += 256) {
    int h = i >> 6, j = i & 63;
    h1[h][j] = fmaxf((h1[h][j] - mv[h][0]) * mv[h][1] * lng[j] + lnb[j], 0.f);
  }
  __syncthreads();
  for (int i = tid; i < H * 32; i += 256) {
    int h = i >> 5, j = i & 31;
    float s = b2[j];
    for (int kk = 0; kk < 64; kk++) s += h1[h][kk] * W2[kk * 32 + j];
    h2[h][j] = fmaxf(s, 0.f);
  }
  __syncthreads();
  if (tid < H) {
    float l[3];
    for (int t = 0; t < 3; t++) {
      float s = b3[t];
      for (int j = 0; j < 32; j++) s += h2[tid][j] * W3[j * 3 + t];
      l[t] = s;
    }
    float mx = fmaxf(l[0], fmaxf(l[1], l[2]));
    float e0 = expf(l[0] - mx), e1 = expf(l[1] - mx), e2 = expf(l[2] - mx);
    float is = 1.f / (e0 + e1 + e2);
    float wt = fminf(fmaxf(wtemp[0], 0.1f), 20.f);
    l[0] = e0 * is / wt; l[1] = e1 * is / wt; l[2] = e2 * is / wt;
    mx = fmaxf(l[0], fmaxf(l[1], l[2]));
    e0 = expf(l[0] - mx); e1 = expf(l[1] - mx); e2 = expf(l[2] - mx);
    is = 1.f / (e0 + e1 + e2);
    float w0 = fminf(fmaxf(e0 * is, 0.05f), 0.8f);
    float w1 = fminf(fmaxf(e1 * is, 0.05f), 0.8f);
    float w2 = fminf(fmaxf(e2 * is, 0.05f), 0.8f);
    is = 1.f / (w0 + w1 + w2);
    SC[36 + tid * 3 + 0] = w0 * is;
    SC[36 + tid * 3 + 1] = w1 * is;
    SC[36 + tid * 3 + 2] = w2 * is;
  }
}

// ---------------- pass 1: cos/cov stats + margin row means ------------------
__global__ void k_sim_pass1(const unsigned short* __restrict__ FQ,
                            const unsigned short* __restrict__ FK,
                            const float* N1I, const float* QDI, const float* MQv,
                            const float* AVv, const float* N2I, const float* KDI,
                            const float* KSUMv, float* SC, float* MROW) {
  __shared__ float rn1[16], rqd[16], rmq[16], ra[16];
  __shared__ float mpart[4][16];
  __shared__ float blk[4];
  int tid = threadIdx.x, wave = tid >> 5, lane = tid & 31;
  int n16 = lane & 15, hi = lane >> 4;
  int hq = blockIdx.x >> 6;
  int sn = (blockIdx.x & 63) * 16;
  int rowbase = hq * NTOK + sn;
  if (tid < 16) {
    rn1[tid] = N1I[rowbase + tid]; rqd[tid] = QDI[rowbase + tid];
    rmq[tid] = MQv[rowbase + tid]; ra[tid]  = AVv[rowbase + tid];
  }
  if (tid < 4) blk[tid] = 0.f;
  float cc = SC[64 + hq];
  const unsigned short* fqp = FQ + (size_t)hq * NTOK * DHEAD;
  const unsigned short* fkp = FK + (size_t)hq * NTOK * DHEAD;
  const unsigned short* arow = fqp + (size_t)(sn + n16) * DHEAD;
  Frag a0, a1;
  a0.h[0] = *(const v8bf*)(arow + 16 * hi);
  a0.h[1] = *(const v8bf*)(arow + 8 + 16 * hi);
  a1.h[0] = *(const v8bf*)(arow + 32 + 16 * hi);
  a1.h[1] = *(const v8bf*)(arow + 40 + 16 * hi);
  __syncthreads();
  float s_cos = 0.f, s_cos2 = 0.f, s_cov = 0.f, s_cov2 = 0.f;
  float mp[8];
#pragma unroll
  for (int r = 0; r < 8; r++) mp[r] = 0.f;
  const float RS = 1.f / (8.f + 1e-6f);
  for (int t = wave; t < 64; t += 4) {
    int mb = t * 16;
    const unsigned short* brow = fkp + (size_t)(mb + n16) * DHEAD;
    if (t + 4 < 64)
      __builtin_prefetch(fkp + (size_t)(mb + 64 + n16) * DHEAD, 0, 3);
    Frag b0, b1;
    b0.v = *(const v16bf*)(brow + 16 * hi);
    b1.v = *(const v16bf*)(brow + 32 + 16 * hi);
    v8f c = {};
    c = wmma_bf(a0, b0, c);
    c = wmma_bf(a1, b1, c);
    int mcol = hq * NTOK + mb + n16;
    float n2i = N2I[mcol], kdi = KDI[mcol], ks = KSUMv[mcol];
#pragma unroll
    for (int r = 0; r < 8; r++) {
      int rl = r + 8 * hi;
      float dot = c[r];
      float cosv = fminf(fmaxf(dot * rn1[rl] * n2i, -0.99f), 0.99f);
      float covp = (dot - ra[rl] - rmq[rl] * ks + rmq[rl] * cc) * RS;
      float csim = fminf(fmaxf(dot * rqd[rl] * kdi, -0.99f), 0.99f);
      float marg = fminf(fmaxf(GAMMA_C - csim, 0.f), 10.f);
      s_cos += cosv; s_cos2 += cosv * cosv;
      s_cov += covp; s_cov2 += covp * covp;
      mp[r] += marg;
    }
  }
  s_cos = wave_sum(s_cos); s_cos2 = wave_sum(s_cos2);
  s_cov = wave_sum(s_cov); s_cov2 = wave_sum(s_cov2);
  if (lane == 0) {
    atomicAdd(&blk[0], s_cos); atomicAdd(&blk[1], s_cos2);
    atomicAdd(&blk[2], s_cov); atomicAdd(&blk[3], s_cov2);
  }
#pragma unroll
  for (int r = 0; r < 8; r++) mp[r] = half_sum(mp[r]);
  if (n16 == 0) {
#pragma unroll
    for (int r = 0; r < 8; r++) mpart[wave][8 * hi + r] = mp[r];
  }
  __syncthreads();
  if (tid < 16) {
    float tot = mpart[0][tid] + mpart[1][tid] + mpart[2][tid] + mpart[3][tid];
    MROW[rowbase + tid] = tot * (1.f / NTOK);
  }
  if (tid < 4) atomicAdd(&SC[tid], blk[tid]);
}

// ---------------- scalars after pass 1 --------------------------------------
__global__ void k_scalars1(const float* MROW, float* SC) {
  __shared__ float aa[256], bb[256];
  int tid = threadIdx.x;
  float s = 0.f, s2 = 0.f;
  for (int i = tid; i < HQ * NTOK; i += 256) { float x = MROW[i]; s += x; s2 += x * x; }
  aa[tid] = s; bb[tid] = s2; __syncthreads();
  for (int off = 128; off; off >>= 1) {
    if (tid < off) { aa[tid] += aa[tid + off]; bb[tid] += bb[tid + off]; }
    __syncthreads();
  }
  if (tid == 0) {
    float cnt = (float)HQ * (float)NTOK * (float)NTOK;
    float vs = (float)NTOK * aa[0], vs2 = (float)NTOK * bb[0];
    float vn = sqrtf(fmaxf((vs2 - vs * vs / cnt) / (cnt - 1.f), 0.f)) + 1e-6f;
    float cs = SC[0], cs2 = SC[1];
    float cn = sqrtf(fmaxf((cs2 - cs * cs / cnt) / (cnt - 1.f), 0.f)) + 1e-6f;
    float cscale = (cn < 1e-4f) ? 0.1f : 1.0f;
    float os = SC[2], os2 = SC[3];
    float covstd = sqrtf(fmaxf((os2 - os * os / cnt) / (cnt - 1.f), 0.f));
    float base = 0.001f / (float)NTOK;
    float reg = (covstd < 1e-6f) ? base * 10.f : base;
    float covn = reg * covstd + 1e-6f;   // clip(+-50) inert at reg~1e-6
    SC[8] = reg;
    for (int h = 0; h < H; h++) {
      SC[12 + h] = SC[36 + h * 3 + 0] * cscale / cn;
      SC[20 + h] = SC[36 + h * 3 + 1] * 0.5f / covn;
      SC[28 + h] = SC[36 + h * 3 + 2] * 0.5f / vn;
    }
  }
}

// ---------------- pass 2: d2 stats + row max --------------------------------
__global__ void k_sim_pass2(const unsigned short* __restrict__ FQ,
                            const unsigned short* __restrict__ FK,
                            const float* N1I, const float* MQv, const float* AVv,
                            const float* N2I, const float* KSUMv,
                            const float* MROW, float* SC, float* RMAX) {
  __shared__ float rn1[16], rmq[16], ra[16], rmr[16];
  __shared__ float mpart[4][16];
  __shared__ float blk[2];
  int tid = threadIdx.x, wave = tid >> 5, lane = tid & 31;
  int n16 = lane & 15, hi = lane >> 4;
  int hq = blockIdx.x >> 6, h = hq >> 2;
  int sn = (blockIdx.x & 63) * 16;
  int rowbase = hq * NTOK + sn;
  if (tid < 16) {
    rn1[tid] = N1I[rowbase + tid]; rmq[tid] = MQv[rowbase + tid];
    ra[tid]  = AVv[rowbase + tid]; rmr[tid] = MROW[rowbase + tid];
  }
  if (tid < 2) blk[tid] = 0.f;
  float cc = SC[64 + hq], reg = SC[8];
  float cA = SC[12 + h], cB = SC[20 + h], cC = SC[28 + h];
  const unsigned short* fqp = FQ + (size_t)hq * NTOK * DHEAD;
  const unsigned short* fkp = FK + (size_t)hq * NTOK * DHEAD;
  const unsigned short* arow = fqp + (size_t)(sn + n16) * DHEAD;
  Frag a0, a1;
  a0.h[0] = *(const v8bf*)(arow + 16 * hi);
  a0.h[1] = *(const v8bf*)(arow + 8 + 16 * hi);
  a1.h[0] = *(const v8bf*)(arow + 32 + 16 * hi);
  a1.h[1] = *(const v8bf*)(arow + 40 + 16 * hi);
  __syncthreads();
  float s_d = 0.f, s_d2 = 0.f;
  float mx[8];
#pragma unroll
  for (int r = 0; r < 8; r++) mx[r] = -3.4e38f;
  const float RS = 1.f / (8.f + 1e-6f);
  for (int t = wave; t < 64; t += 4) {
    int mb = t * 16;
    const unsigned short* brow = fkp + (size_t)(mb + n16) * DHEAD;
    if (t + 4 < 64)
      __builtin_prefetch(fkp + (size_t)(mb + 64 + n16) * DHEAD, 0, 3);
    Frag b0, b1;
    b0.v = *(const v16bf*)(brow + 16 * hi);
    b1.v = *(const v16bf*)(brow + 32 + 16 * hi);
    v8f c = {};
    c = wmma_bf(a0, b0, c);
    c = wmma_bf(a1, b1, c);
    int mcol = hq * NTOK + mb + n16;
    float n2i = N2I[mcol], ks = KSUMv[mcol];
#pragma unroll
    for (int r = 0; r < 8; r++) {
      int rl = r + 8 * hi;
      float dot = c[r];
      float cosv = fminf(fmaxf(dot * rn1[rl] * n2i, -0.99f), 0.99f);
      float covp = (dot - ra[rl] - rmq[rl] * ks + rmq[rl] * cc) * RS;
      float cove = fminf(fmaxf(reg * covp, -50.f), 50.f);
      float d2 = cA * cosv + cB * cove + cC * rmr[rl];
      s_d += d2; s_d2 += d2 * d2;
      mx[r] = fmaxf(mx[r], d2);
    }
  }
  s_d = wave_sum(s_d); s_d2 = wave_sum(s_d2);
  if (lane == 0) { atomicAdd(&blk[0], s_d); atomicAdd(&blk[1], s_d2); }
#pragma unroll
  for (int r = 0; r < 8; r++) mx[r] = half_max(mx[r]);
  if (n16 == 0) {
#pragma unroll
    for (int r = 0; r < 8; r++) mpart[wave][8 * hi + r] = mx[r];
  }
  __syncthreads();
  if (tid < 16)
    RMAX[rowbase + tid] = fmaxf(fmaxf(mpart[0][tid], mpart[1][tid]),
                                fmaxf(mpart[2][tid], mpart[3][tid]));
  if (tid < 2) atomicAdd(&SC[4 + tid], blk[tid]);
}

__global__ void k_scalars2(float* SC) {
  if (threadIdx.x == 0) {
    float cnt = (float)HQ * (float)NTOK * (float)NTOK;
    float s = SC[4], s2 = SC[5];
    float dstd = sqrtf(fmaxf((s2 - s * s / cnt) / (cnt - 1.f), 0.f));
    float temp = (dstd < 1e-6f) ? 0.1f : 0.3f + dstd;
    SC[9] = 1.f / fminf(fmaxf(temp, 0.1f), 5.f);
  }
}

// ---------------- pass 3: softmax(d2/T) @ V ---------------------------------
__global__ void __launch_bounds__(128) k_sim_pass3(
    const unsigned short* __restrict__ FQ, const unsigned short* __restrict__ FK,
    const unsigned short* __restrict__ FVT,
    const float* N1I, const float* MQv, const float* AVv,
    const float* N2I, const float* KSUMv, const float* MROW, const float* RMAX,
    const float* SC, unsigned short* __restrict__ OH) {
  __shared__ unsigned short P[16 * 1024];     // 32 KB strip of exp scores (bf16)
  __shared__ float rn1[16], rmq[16], ra[16], rmr[16], rmx[16], rsinv[16];
  __shared__ float spart[4][16];
  int tid = threadIdx.x, wave = tid >> 5, lane = tid & 31;
  int n16 = lane & 15, hi = lane >> 4;
  int hq = blockIdx.x >> 6, h = hq >> 2, q = hq & 3;
  int sn = (blockIdx.x & 63) * 16;
  int rowbase = hq * NTOK + sn;
  if (tid < 16) {
    rn1[tid] = N1I[rowbase + tid]; rmq[tid] = MQv[rowbase + tid];
    ra[tid]  = AVv[rowbase + tid]; rmr[tid] = MROW[rowbase + tid];
    rmx[tid] = RMAX[rowbase + tid];
  }
  float cc = SC[64 + hq], reg = SC[8], invt = SC[9];
  float cA = SC[12 + h], cB = SC[20 + h], cC = SC[28 + h];
  const unsigned short* fqp = FQ + (size_t)hq * NTOK * DHEAD;
  const unsigned short* fkp = FK + (size_t)hq * NTOK * DHEAD;
  const unsigned short* arow = fqp + (size_t)(sn + n16) * DHEAD;
  Frag a0, a1;
  a0.h[0] = *(const v8bf*)(arow + 16 * hi);
  a0.h[1] = *(const v8bf*)(arow + 8 + 16 * hi);
  a1.h[0] = *(const v8bf*)(arow + 32 + 16 * hi);
  a1.h[1] = *(const v8bf*)(arow + 40 + 16 * hi);
  __syncthreads();
  float sp[8];
#pragma unroll
  for (int r = 0; r < 8; r++) sp[r] = 0.f;
  const float RS = 1.f / (8.f + 1e-6f);
  for (int t = wave; t < 64; t += 4) {
    int mb = t * 16;
    const unsigned short* brow = fkp + (size_t)(mb + n16) * DHEAD;
    if (t + 4 < 64)
      __builtin_prefetch(fkp + (size_t)(mb + 64 + n16) * DHEAD, 0, 3);
    Frag b0, b1;
    b0.v = *(const v16bf*)(brow + 16 * hi);
    b1.v = *(const v16bf*)(brow + 32 + 16 * hi);
    v8f c = {};
    c = wmma_bf(a0, b0, c);
    c = wmma_bf(a1, b1, c);
    int mcol = hq * NTOK + mb + n16;
    float n2i = N2I[mcol], ks = KSUMv[mcol];
#pragma unroll
    for (int r = 0; r < 8; r++) {
      int rl = r + 8 * hi;
      float dot = c[r];
      float cosv = fminf(fmaxf(dot * rn1[rl] * n2i, -0.99f), 0.99f);
      float covp = (dot - ra[rl] - rmq[rl] * ks + rmq[rl] * cc) * RS;
      float cove = fminf(fmaxf(reg * covp, -50.f), 50.f);
      float d2 = cA * cosv + cB * cove + cC * rmr[rl];
      float p = exp2f((d2 - rmx[rl]) * invt * 1.44269504f);
      sp[r] += p;
      P[rl * 1024 + mb + n16] = f2bf(p);
    }
  }
#pragma unroll
  for (int r = 0; r < 8; r++) sp[r] = half_sum(sp[r]);
  if (n16 == 0) {
#pragma unroll
    for (int r = 0; r < 8; r++) spart[wave][8 * hi + r] = sp[r];
  }
  __syncthreads();
  if (tid < 16)
    rsinv[tid] = 1.f / (spart[0][tid] + spart[1][tid] + spart[2][tid] + spart[3][tid]);
  __syncthreads();
  // P[16x1024] @ fv[1024x64]; each wave owns one 16-wide d block
  int db = wave * 16;
  const unsigned short* fvp = FVT + (size_t)hq * DHEAD * NTOK + (size_t)(db + n16) * NTOK;
  v8f c = {};
#pragma unroll 4
  for (int kb = 0; kb < NTOK; kb += 32) {
    Frag a, b;
    const unsigned short* pr = &P[n16 * 1024 + kb];
    if (kb + 32 < NTOK)
      __builtin_prefetch(fvp + kb + 32 + 16 * hi, 0, 3);
    a.h[0] = *(const v8bf*)(pr + 16 * hi);
    a.h[1] = *(const v8bf*)(pr + 8 + 16 * hi);
    b.v    = *(const v16bf*)(fvp + kb + 16 * hi);
    c = wmma_bf(a, b, c);
  }
#pragma unroll
  for (int r = 0; r < 8; r++) {
    int rl = r + 8 * hi;
    int n = sn + rl;
    float o = c[r] * rsinv[rl];
    OH[(size_t)(q * NTOK + n) * INNER + h * DHEAD + db + n16] = f2bf(o);
  }
}

// ---------------- output projection: OH[4096x512] @ WoutT + b ---------------
__global__ void k_out_gemm(const unsigned short* __restrict__ OH,
                           const unsigned short* __restrict__ WT,
                           const float* __restrict__ bout, float* __restrict__ OUT) {
  int wave = threadIdx.x >> 5, lane = threadIdx.x & 31;
  int n16 = lane & 15, hi = lane >> 4;
  int tile = blockIdx.x * 4 + wave;
  int mt = tile >> 5, jt = tile & 31;
  const unsigned short* ar = OH + (size_t)(mt * 16 + n16) * INNER;
  const unsigned short* br = WT + (size_t)(jt * 16 + n16) * INNER;
  v8f c = {};
#pragma unroll 4
  for (int kb = 0; kb < INNER; kb += 32) {
    Frag a, b;
    a.h[0] = *(const v8bf*)(ar + kb + 16 * hi);
    a.h[1] = *(const v8bf*)(ar + kb + 8 + 16 * hi);
    b.v    = *(const v16bf*)(br + kb + 16 * hi);
    c = wmma_bf(a, b, c);
  }
  float bb = bout[jt * 16 + n16];
#pragma unroll
  for (int r = 0; r < 8; r++)
    OUT[(size_t)(mt * 16 + r + 8 * hi) * DIM + jt * 16 + n16] = c[r] + bb;
}

// ---------------------------------------------------------------------------
extern "C" void kernel_launch(void* const* d_in, const int* in_sizes, int n_in,
                              void* d_out, int out_size, void* d_ws, size_t ws_size,
                              hipStream_t stream) {
  (void)in_sizes; (void)n_in; (void)out_size; (void)ws_size;
  const float* q      = (const float*)d_in[0];
  const float* k      = (const float*)d_in[1];
  const float* v      = (const float*)d_in[2];
  const float* lng    = (const float*)d_in[3];
  const float* lnb    = (const float*)d_in[4];
  const float* W_in   = (const float*)d_in[5];
  const float* wpW1   = (const float*)d_in[6];
  const float* wpb1   = (const float*)d_in[7];
  const float* wplng  = (const float*)d_in[8];
  const float* wplnb  = (const float*)d_in[9];
  const float* wpW2   = (const float*)d_in[10];
  const float* wpb2   = (const float*)d_in[11];
  const float* wpW3   = (const float*)d_in[12];
  const float* wpb3   = (const float*)d_in[13];
  const float* wtemp  = (const float*)d_in[14];
  const float* W_out  = (const float*)d_in[15];
  const float* b_out  = (const float*)d_in[16];

  char* p = (char*)d_ws;
  auto alloc = [&](size_t bytes) -> char* {
    char* r = p; p += (bytes + 255) & ~(size_t)255; return r;
  };
  float* SC    = (float*)alloc(256 * 4);
  float* QGS   = (float*)alloc(512 * 4);
  float* KGS   = (float*)alloc(512 * 4);
  float* MKB   = (float*)alloc(HQ * DHEAD * 4);
  float* N1I   = (float*)alloc(HQ * NTOK * 4);
  float* QDI   = (float*)alloc(HQ * NTOK * 4);
  float* MQv   = (float*)alloc(HQ * NTOK * 4);
  float* AVv   = (float*)alloc(HQ * NTOK * 4);
  float* N2I   = (float*)alloc(HQ * NTOK * 4);
  float* KDI   = (float*)alloc(HQ * NTOK * 4);
  float* KSUMv = (float*)alloc(HQ * NTOK * 4);
  float* MROW  = (float*)alloc(HQ * NTOK * 4);
  float* RMAX  = (float*)alloc(HQ * NTOK * 4);
  unsigned short* WinT  = (unsigned short*)alloc((size_t)DIM * INNER * 2);
  unsigned short* WoutT = (unsigned short*)alloc((size_t)INNER * DIM * 2);
  unsigned short* XLNQ  = (unsigned short*)alloc((size_t)NROWS * DIM * 2);
  unsigned short* XLNK  = (unsigned short*)alloc((size_t)NROWS * DIM * 2);
  unsigned short* XLNV  = (unsigned short*)alloc((size_t)NROWS * DIM * 2);
  unsigned short* FQ    = (unsigned short*)alloc((size_t)HQ * NTOK * DHEAD * 2);
  unsigned short* FK    = (unsigned short*)alloc((size_t)HQ * NTOK * DHEAD * 2);
  unsigned short* FVT   = (unsigned short*)alloc((size_t)HQ * NTOK * DHEAD * 2);
  unsigned short* OH    = (unsigned short*)alloc((size_t)NROWS * INNER * 2);

  k_zero<<<1, 256, 0, stream>>>(SC, QGS, KGS);
  k_transpose_w<<<1024, 256, 0, stream>>>(W_in, WinT);
  k_transpose_w<<<1024, 256, 0, stream>>>(W_out, WoutT);
  k_layernorm<<<1024, 128, 0, stream>>>(q, lng, lnb, XLNQ);
  k_layernorm<<<1024, 128, 0, stream>>>(k, lng, lnb, XLNK);
  k_layernorm<<<1024, 128, 0, stream>>>(v, lng, lnb, XLNV);
  k_proj_gemm<<<2048, 128, 0, stream>>>(XLNQ, WinT, FQ, FK, FVT, 0);
  k_proj_gemm<<<2048, 128, 0, stream>>>(XLNK, WinT, FQ, FK, FVT, 1);
  k_proj_gemm<<<2048, 128, 0, stream>>>(XLNV, WinT, FQ, FK, FVT, 2);
  k_rowstats<<<HQ, 256, 0, stream>>>(FQ, N1I, QDI, MQv, QGS, MKB, 0);
  k_rowstats<<<HQ, 256, 0, stream>>>(FK, N2I, KDI, KSUMv, KGS, MKB, 1);
  k_compute_a<<<HQ * NTOK / 4, 128, 0, stream>>>(FQ, MKB, AVv);
  k_mlp<<<1, 256, 0, stream>>>(QGS, KGS, MKB, wpW1, wpb1, wplng, wplnb,
                               wpW2, wpb2, wpW3, wpb3, wtemp, SC);
  k_sim_pass1<<<HQ * 64, 128, 0, stream>>>(FQ, FK, N1I, QDI, MQv, AVv,
                                           N2I, KDI, KSUMv, SC, MROW);
  k_scalars1<<<1, 256, 0, stream>>>(MROW, SC);
  k_sim_pass2<<<HQ * 64, 128, 0, stream>>>(FQ, FK, N1I, MQv, AVv,
                                           N2I, KSUMv, MROW, SC, RMAX);
  k_scalars2<<<1, 32, 0, stream>>>(SC);
  k_sim_pass3<<<HQ * 64, 128, 0, stream>>>(FQ, FK, FVT, N1I, MQv, AVv,
                                           N2I, KSUMv, MROW, RMAX, SC, OH);
  k_out_gemm<<<2048, 128, 0, stream>>>(OH, WoutT, b_out, (float*)d_out);
}
